// Hydra_Block_18957985644896
// MI455X (gfx1250) — compile-verified, matrix operand order checked by reading, not run on hardware
//
#include <hip/hip_runtime.h>

// ---------------------------------------------------------------- types ----
typedef __bf16 bf16_t;
typedef __attribute__((ext_vector_type(16))) __bf16 bf16x16;
typedef __attribute__((ext_vector_type(8)))  __bf16 bf16x8;
typedef __attribute__((ext_vector_type(8)))  float  f32x8;

#define DEV __device__ __forceinline__

// ----------------------------------------------------------- dimensions ----
constexpr int DM   = 1024;   // d_model
constexpr int DI   = 2048;   // d_inner
constexpr int DSTATE = 64;
constexpr int NH   = 32;     // heads
constexpr int HD   = 64;     // head dim
constexpr int DIP  = 4288;   // in_proj rows
constexpr int CDIM = 2176;   // conv dim
constexpr int DCONV = 4;
constexpr int CHK  = 64;     // chunk
constexpr int NC   = 32;     // chunks per seq
constexpr int NB   = 4;      // batch
constexpr int SL   = 2048;   // seqlen
constexpr int NB2  = 8;      // 2*batch (fwd+bwd)
constexpr float EPSF = 1e-5f;

// ------------------------------------------------------- ws layout (B) ----
constexpr size_t OFF_HBF  = 0;                                    // 8192*1024 bf16
constexpr size_t OFF_WIN  = OFF_HBF  + (size_t)NB*SL*DM*2;        // 4288*1024 bf16
constexpr size_t OFF_WOUT = OFF_WIN  + (size_t)DIP*DM*2;          // 1024*2048 bf16
constexpr size_t OFF_ZX   = OFF_WOUT + (size_t)DM*DI*2;           // 8192*4288 f32
constexpr size_t OFF_DT2  = OFF_ZX   + (size_t)NB*SL*DIP*4;       // 8*2048*32 f32
constexpr size_t OFF_DA   = OFF_DT2  + (size_t)NB2*SL*NH*4;       // 8*2048*32 f32
constexpr size_t OFF_XCV  = OFF_DA   + (size_t)NB2*SL*NH*4;       // 4*2048*2048 f32
constexpr size_t OFF_X2   = OFF_XCV  + (size_t)NB*SL*DI*4;        // 8*2048*2048 bf16
constexpr size_t OFF_B2   = OFF_X2   + (size_t)NB2*SL*DI*2;       // 8*2048*64 bf16
constexpr size_t OFF_C2   = OFF_B2   + (size_t)NB2*SL*DSTATE*2;   // 8*2048*64 bf16
constexpr size_t OFF_ST   = OFF_C2   + (size_t)NB2*SL*DSTATE*2;   // 8*32*32*4096 f32
constexpr size_t OFF_AC   = OFF_ST   + (size_t)NB2*NH*NC*HD*DSTATE*4; // 8*32*32*64 f32
constexpr size_t OFF_Y2   = OFF_AC   + (size_t)NB2*NH*NC*CHK*4;   // 8*2048*2048 f32
constexpr size_t OFF_YN   = OFF_Y2   + (size_t)NB2*SL*DI*4;       // 8192*2048 bf16

// ---------------------------------------------------------- wmma utils ----
DEV f32x8 fzero() {
  f32x8 v;
#pragma unroll
  for (int i = 0; i < 8; ++i) v[i] = 0.f;
  return v;
}

DEV f32x8 wmma_bf(bf16x16 a, bf16x16 b, f32x8 c) {
#if defined(__HIP_DEVICE_COMPILE__)
  return __builtin_amdgcn_wmma_f32_16x16x32_bf16(false, a, false, b,
                                                 (short)0, c, false, false);
#else
  (void)a; (void)b; return c;
#endif
}

// Load one 16-bit A/B operand fragment from an LDS tile.
// Row-major tile, `ld` elements per row (16B-aligned stride).
// Per ISA 16-bit layout: lane half h=lane/16 supplies k = h*8..h*8+7 and
// 16+h*8..16+h*8+7 within the 32-wide k-slab -> two 16-byte groups.
DEV bf16x16 ld_frag(const bf16_t* base, int row, int ld, int kslab, int lane) {
  const bf16_t* p = base + row * ld + kslab * 32 + ((lane >> 4) << 3);
  bf16x8 lo = *(const bf16x8*)p;
  bf16x8 hi = *(const bf16x8*)(p + 16);
  return __builtin_shufflevector(lo, hi, 0,1,2,3,4,5,6,7,8,9,10,11,12,13,14,15);
}

// ------------------------------------------------------------- kernels ----

// f32 -> bf16 elementwise
__global__ __launch_bounds__(256) void cvt_bf16_kernel(const float* __restrict__ src,
                                                       bf16_t* __restrict__ dst, long n) {
  long i = (long)blockIdx.x * blockDim.x + threadIdx.x;
  long stride = (long)gridDim.x * blockDim.x;
  for (; i < n; i += stride) dst[i] = (bf16_t)src[i];
}

// res = hs + residual  (-> second output); h = rmsnorm(res)*w -> bf16
__global__ __launch_bounds__(256) void addnorm_kernel(const float* __restrict__ hs,
                                                      const float* __restrict__ resid,
                                                      const float* __restrict__ nw,
                                                      float* __restrict__ res_out,
                                                      bf16_t* __restrict__ hbf) {
  int t = blockIdx.x, b = blockIdx.y, tid = threadIdx.x;
  size_t row = (size_t)b * SL + t;
  __shared__ float xsh[DM];
  __shared__ float rbuf[256];
  float ss = 0.f;
  for (int i = tid; i < DM; i += 256) {
    float v = hs[row * DM + i] + resid[row * DM + i];
    xsh[i] = v; res_out[row * DM + i] = v; ss += v * v;
  }
  rbuf[tid] = ss; __syncthreads();
  for (int s = 128; s > 0; s >>= 1) { if (tid < s) rbuf[tid] += rbuf[tid + s]; __syncthreads(); }
  float rs = rsqrtf(rbuf[0] / DM + EPSF);
  for (int i = tid; i < DM; i += 256)
    hbf[row * DM + i] = (bf16_t)(xsh[i] * rs * nw[i]);
}

// C[M,N] (f32) = A[M,K](bf16,row-major) @ W[N,K](bf16,row-major)^T
// block tile 128x64, 256 threads = 8 waves (4 M x 2 N), wave tile 32x32.
__global__ __launch_bounds__(256) void gemm_bf16_kernel(const bf16_t* __restrict__ A,
                                                        const bf16_t* __restrict__ W,
                                                        float* __restrict__ C,
                                                        int M, int N, int K) {
  constexpr int LDA = 40;  // 32 + 8 pad, keeps 16B alignment (80B rows)
  __shared__ __align__(16) bf16_t As[128 * LDA];
  __shared__ __align__(16) bf16_t Ws[64 * LDA];
  const int tid = threadIdx.x, lane = tid & 31, wave = tid >> 5;
  const int wm = wave & 3, wn = wave >> 2;
  const long bm = (long)blockIdx.x * 128, bn = (long)blockIdx.y * 64;
  f32x8 acc[2][2];
#pragma unroll
  for (int i = 0; i < 2; ++i)
#pragma unroll
    for (int j = 0; j < 2; ++j) acc[i][j] = fzero();

  const int rowA = tid >> 1, qA = (tid & 1) * 2;  // 2 uint4 per thread
  const int rowW = tid >> 2, qW = tid & 3;        // 1 uint4 per thread

  for (int k0 = 0; k0 < K; k0 += 32) {
    const uint4* srcA = (const uint4*)(A + (bm + rowA) * (long)K + k0);
    uint4 a0 = srcA[qA], a1 = srcA[qA + 1];
    uint4 w0 = ((const uint4*)(W + (bn + rowW) * (long)K + k0))[qW];
    if (k0 + 32 < K) {
      __builtin_prefetch(A + (bm + rowA) * (long)K + k0 + 32, 0, 1);
      __builtin_prefetch(W + (bn + rowW) * (long)K + k0 + 32, 0, 1);
    }
    uint4* dA = (uint4*)(As + rowA * LDA);
    dA[qA] = a0; dA[qA + 1] = a1;
    ((uint4*)(Ws + rowW * LDA))[qW] = w0;
    __syncthreads();

    bf16x16 af[2], bf[2];
#pragma unroll
    for (int mt = 0; mt < 2; ++mt)
      af[mt] = ld_frag(As, wm * 32 + mt * 16 + (lane & 15), LDA, 0, lane);
#pragma unroll
    for (int nt = 0; nt < 2; ++nt)
      bf[nt] = ld_frag(Ws, wn * 32 + nt * 16 + (lane & 15), LDA, 0, lane);
#pragma unroll
    for (int mt = 0; mt < 2; ++mt)
#pragma unroll
      for (int nt = 0; nt < 2; ++nt)
        acc[mt][nt] = wmma_bf(af[mt], bf[nt], acc[mt][nt]);
    __syncthreads();
  }
#pragma unroll
  for (int mt = 0; mt < 2; ++mt)
#pragma unroll
    for (int nt = 0; nt < 2; ++nt)
#pragma unroll
      for (int r = 0; r < 8; ++r) {
        long m = bm + wm * 32 + mt * 16 + r + ((lane >> 4) << 3);
        long n = bn + wn * 32 + nt * 16 + (lane & 15);
        C[m * (long)N + n] = acc[mt][nt][r];
      }
}

DEV float softplusf(float x) { return x > 20.f ? x : log1pf(__expf(x)); }

// dt2[d,t,h] = softplus(dt_raw + bias); dA = dt2 * (-exp(A_log))
__global__ __launch_bounds__(64) void dt_kernel(const float* __restrict__ zx,
                                                const float* __restrict__ dt_bias,
                                                const float* __restrict__ A_log,
                                                float* __restrict__ dt2,
                                                float* __restrict__ dA) {
  int t = blockIdx.x, b = blockIdx.y, tid = threadIdx.x;
  size_t row = (size_t)b * SL + t;
  int h = tid & 31;
  float Ah = -__expf(A_log[h]);
  if (tid < 32) {  // forward direction
    float v = softplusf(zx[row * DIP + (DI + CDIM) + h] + dt_bias[h]);
    size_t o = ((size_t)b * SL + t) * NH + h;
    dt2[o] = v; dA[o] = v * Ah;
  } else {         // backward direction, time-flipped
    float v = softplusf(zx[row * DIP + (DI + CDIM) + NH + h] + dt_bias[h]);
    size_t o = ((size_t)(b + NB) * SL + (SL - 1 - t)) * NH + h;
    dt2[o] = v; dA[o] = v * Ah;
  }
}

// causal depthwise conv(4) + silu; split into x / B / C; emit fwd+bwd copies
// X2 = silu(conv(x)) * dt2 (both dirs, bf16); xconv fwd f32 kept for Deff.
__global__ __launch_bounds__(256) void conv_kernel(const float* __restrict__ zx,
                                                   const float* __restrict__ conv_w,
                                                   const float* __restrict__ conv_b,
                                                   const float* __restrict__ dt2,
                                                   float* __restrict__ xconv,
                                                   bf16_t* __restrict__ X2,
                                                   bf16_t* __restrict__ B2,
                                                   bf16_t* __restrict__ C2) {
  int t = blockIdx.x, b = blockIdx.y, tid = threadIdx.x;
  int tb = SL - 1 - t;
  size_t row = (size_t)b * SL + t;
  for (int c = tid; c < CDIM; c += 256) {
    float v = conv_b[c];
#pragma unroll
    for (int k = 0; k < DCONV; ++k) {
      int tt = t - (DCONV - 1) + k;
      if (tt >= 0) v += zx[((size_t)b * SL + tt) * DIP + DI + c] * conv_w[c * DCONV + k];
    }
    v = v / (1.f + __expf(-v));  // silu
    if (c < DI) {
      int h = c >> 6, p = c & 63;
      xconv[row * DI + c] = v;
      float df = dt2[((size_t)b * SL + t) * NH + h];
      float db = dt2[((size_t)(b + NB) * SL + tb) * NH + h];
      X2[(((size_t)b * SL + t) * NH + h) * HD + p] = (bf16_t)(v * df);
      X2[(((size_t)(b + NB) * SL + tb) * NH + h) * HD + p] = (bf16_t)(v * db);
    } else if (c < DI + DSTATE) {
      int n = c - DI;
      B2[((size_t)b * SL + t) * DSTATE + n] = (bf16_t)v;
      B2[((size_t)(b + NB) * SL + tb) * DSTATE + n] = (bf16_t)v;
    } else {
      int n = c - DI - DSTATE;
      C2[((size_t)b * SL + t) * DSTATE + n] = (bf16_t)v;
      C2[((size_t)(b + NB) * SL + tb) * DSTATE + n] = (bf16_t)v;
    }
  }
}

// One wave per (chunk, head, dir): Y_diag and per-chunk states via WMMA.
__global__ __launch_bounds__(32) void ssd_chunk_kernel(const bf16_t* __restrict__ X2,
                                                       const bf16_t* __restrict__ B2,
                                                       const bf16_t* __restrict__ C2,
                                                       const float* __restrict__ dA,
                                                       float* __restrict__ y2,
                                                       float* __restrict__ states,
                                                       float* __restrict__ AcumG) {
  constexpr int LDT = 72;  // 64 + 8 pad (144B rows, 16B aligned)
  const int c = blockIdx.x, h = blockIdx.y, d = blockIdx.z;
  const int lane = threadIdx.x;
  __shared__ __align__(16) bf16_t Cc[64 * LDT];
  __shared__ __align__(16) bf16_t Bc[64 * LDT];
  __shared__ __align__(16) bf16_t Xt[64 * LDT];   // X^T : [p][l]
  __shared__ __align__(16) bf16_t Mm[64 * LDT];   // masked G : [l][s]
  __shared__ __align__(16) bf16_t Bdt[64 * LDT];  // (decay*B)^T : [n][l]
  __shared__ float Acum[64];

  const size_t tbase = (size_t)d * SL + (size_t)c * CHK;  // flattened (d,t) row
  for (int idx = lane; idx < 64 * 64; idx += 32) {
    int l = idx >> 6, n = idx & 63;
    Cc[l * LDT + n] = C2[(tbase + l) * DSTATE + n];
    Bc[l * LDT + n] = B2[(tbase + l) * DSTATE + n];
    // X^T staging: Xt[p][l]
    Xt[n * LDT + l] = X2[((tbase + l) * NH + h) * HD + n];
  }
  if (lane == 0) {
    float s = 0.f;
    for (int l = 0; l < CHK; ++l) { s += dA[(tbase + l) * NH + h]; Acum[l] = s; }
  }
  __syncthreads();
  const float alast = Acum[63];
  for (int idx = lane; idx < 64 * 64; idx += 32) {
    int l = idx >> 6, n = idx & 63;
    Bdt[n * LDT + l] = (bf16_t)((float)Bc[l * LDT + n] * __expf(alast - Acum[l]));
  }
  __syncthreads();

  // ---- G = C @ B^T ----
  f32x8 acc[4][4];
#pragma unroll
  for (int i = 0; i < 4; ++i)
#pragma unroll
    for (int j = 0; j < 4; ++j) acc[i][j] = fzero();
#pragma unroll
  for (int ks = 0; ks < 2; ++ks)
#pragma unroll
    for (int mt = 0; mt < 4; ++mt) {
      bf16x16 a = ld_frag(Cc, mt * 16 + (lane & 15), LDT, ks, lane);
#pragma unroll
      for (int nt = 0; nt < 4; ++nt)
        acc[mt][nt] = wmma_bf(a, ld_frag(Bc, nt * 16 + (lane & 15), LDT, ks, lane),
                              acc[mt][nt]);
    }
  // mask with exp(segsum), write Mm (bf16)
#pragma unroll
  for (int mt = 0; mt < 4; ++mt)
#pragma unroll
    for (int nt = 0; nt < 4; ++nt)
#pragma unroll
      for (int r = 0; r < 8; ++r) {
        int l = mt * 16 + r + ((lane >> 4) << 3);
        int s = nt * 16 + (lane & 15);
        float f = (s <= l) ? __expf(Acum[l] - Acum[s]) : 0.f;
        Mm[l * LDT + s] = (bf16_t)(acc[mt][nt][r] * f);
      }
  __syncthreads();

  // ---- Y_diag = Mm @ X   (B-operand = rows of Xt) ----
#pragma unroll
  for (int i = 0; i < 4; ++i)
#pragma unroll
    for (int j = 0; j < 4; ++j) acc[i][j] = fzero();
#pragma unroll
  for (int ks = 0; ks < 2; ++ks)
#pragma unroll
    for (int mt = 0; mt < 4; ++mt) {
      bf16x16 a = ld_frag(Mm, mt * 16 + (lane & 15), LDT, ks, lane);
#pragma unroll
      for (int nt = 0; nt < 4; ++nt)
        acc[mt][nt] = wmma_bf(a, ld_frag(Xt, nt * 16 + (lane & 15), LDT, ks, lane),
                              acc[mt][nt]);
    }
#pragma unroll
  for (int mt = 0; mt < 4; ++mt)
#pragma unroll
    for (int nt = 0; nt < 4; ++nt)
#pragma unroll
      for (int r = 0; r < 8; ++r) {
        int l = mt * 16 + r + ((lane >> 4) << 3);
        int p = nt * 16 + (lane & 15);
        y2[(tbase + l) * DI + h * HD + p] = acc[mt][nt][r];
      }

  // ---- states[p,n] = sum_l Xt[p,l] * Bdt[n,l] ----
#pragma unroll
  for (int i = 0; i < 4; ++i)
#pragma unroll
    for (int j = 0; j < 4; ++j) acc[i][j] = fzero();
#pragma unroll
  for (int ks = 0; ks < 2; ++ks)
#pragma unroll
    for (int mt = 0; mt < 4; ++mt) {
      bf16x16 a = ld_frag(Xt, mt * 16 + (lane & 15), LDT, ks, lane);
#pragma unroll
      for (int nt = 0; nt < 4; ++nt)
        acc[mt][nt] = wmma_bf(a, ld_frag(Bdt, nt * 16 + (lane & 15), LDT, ks, lane),
                              acc[mt][nt]);
    }
  const size_t sbase = (((size_t)d * NH + h) * NC + c) * (HD * DSTATE);
#pragma unroll
  for (int mt = 0; mt < 4; ++mt)
#pragma unroll
    for (int nt = 0; nt < 4; ++nt)
#pragma unroll
      for (int r = 0; r < 8; ++r) {
        int p = mt * 16 + r + ((lane >> 4) << 3);
        int n = nt * 16 + (lane & 15);
        states[sbase + p * DSTATE + n] = acc[mt][nt][r];
      }
  const size_t abase = (((size_t)d * NH + h) * NC + c) * CHK;
  for (int l = lane; l < CHK; l += 32) AcumG[abase + l] = Acum[l];
}

// Exclusive inter-chunk scan (in place): out[c]=S; S = S*exp(Atot[c]) + cs[c]
__global__ __launch_bounds__(256) void ssd_scan_kernel(float* __restrict__ states,
                                                       const float* __restrict__ AcumG) {
  const int h = blockIdx.x, d = blockIdx.y, tid = threadIdx.x;
  float S[16];
#pragma unroll
  for (int k = 0; k < 16; ++k) S[k] = 0.f;
  const size_t hb = ((size_t)d * NH + h) * NC;
  for (int c = 0; c < NC; ++c) {
    float ex = __expf(AcumG[(hb + c) * CHK + 63]);
    size_t base = (hb + c) * (HD * DSTATE);
#pragma unroll
    for (int k = 0; k < 16; ++k) {
      size_t idx = base + (size_t)k * 256 + tid;
      float cs = states[idx];
      states[idx] = S[k];
      S[k] = S[k] * ex + cs;
    }
  }
}

// Y_off[l,p] = exp(Acum[l]) * sum_n C[l,n] * S[p,n];  y2 += Y_off
__global__ __launch_bounds__(32) void ssd_off_kernel(const bf16_t* __restrict__ C2,
                                                     const float* __restrict__ states,
                                                     const float* __restrict__ AcumG,
                                                     float* __restrict__ y2) {
  constexpr int LDT = 72;
  const int c = blockIdx.x, h = blockIdx.y, d = blockIdx.z;
  const int lane = threadIdx.x;
  __shared__ __align__(16) bf16_t Cc[64 * LDT];
  __shared__ __align__(16) bf16_t Sb[64 * LDT];  // states rows [p][n]
  __shared__ float Acs[64];
  const size_t tbase = (size_t)d * SL + (size_t)c * CHK;
  const size_t sbase = (((size_t)d * NH + h) * NC + c) * (HD * DSTATE);
  const size_t abase = (((size_t)d * NH + h) * NC + c) * CHK;
  for (int idx = lane; idx < 64 * 64; idx += 32) {
    int r0 = idx >> 6, cc = idx & 63;
    Cc[r0 * LDT + cc] = C2[(tbase + r0) * DSTATE + cc];
    Sb[r0 * LDT + cc] = (bf16_t)states[sbase + r0 * DSTATE + cc];
  }
  for (int l = lane; l < CHK; l += 32) Acs[l] = AcumG[abase + l];
  __syncthreads();

  f32x8 acc[4][4];
#pragma unroll
  for (int i = 0; i < 4; ++i)
#pragma unroll
    for (int j = 0; j < 4; ++j) acc[i][j] = fzero();
#pragma unroll
  for (int ks = 0; ks < 2; ++ks)
#pragma unroll
    for (int mt = 0; mt < 4; ++mt) {
      bf16x16 a = ld_frag(Cc, mt * 16 + (lane & 15), LDT, ks, lane);
#pragma unroll
      for (int nt = 0; nt < 4; ++nt)
        acc[mt][nt] = wmma_bf(a, ld_frag(Sb, nt * 16 + (lane & 15), LDT, ks, lane),
                              acc[mt][nt]);
    }
#pragma unroll
  for (int mt = 0; mt < 4; ++mt)
#pragma unroll
    for (int nt = 0; nt < 4; ++nt)
#pragma unroll
      for (int r = 0; r < 8; ++r) {
        int l = mt * 16 + r + ((lane >> 4) << 3);
        int p = nt * 16 + (lane & 15);
        size_t idx = (tbase + l) * DI + h * HD + p;
        y2[idx] += __expf(Acs[l]) * acc[mt][nt][r];
      }
}

// roll+flip combine, Deff (fc_D), skip, gate with silu(z), rmsnorm -> bf16
__global__ __launch_bounds__(256) void post_kernel(const float* __restrict__ y2,
                                                   const float* __restrict__ xconv,
                                                   const float* __restrict__ zx,
                                                   const float* __restrict__ fcw,
                                                   const float* __restrict__ Dv,
                                                   const float* __restrict__ gnw,
                                                   bf16_t* __restrict__ yn) {
  int t = blockIdx.x, b = blockIdx.y, tid = threadIdx.x;
  int tb = SL - 1 - t;
  size_t row = (size_t)b * SL + t;
  __shared__ float xsh[DI];
  __shared__ float gsh[DI];
  __shared__ float part[8 * NH];
  __shared__ float defs[NH];
  __shared__ float rbuf[256];
  for (int i = tid; i < DI; i += 256) xsh[i] = xconv[row * DI + i];
  __syncthreads();
  { int h = tid & 31, pp = tid >> 5;
    const float* wrow = fcw + (size_t)h * DI + pp * 256;
    const float* xp = xsh + pp * 256;
    float s = 0.f;
    for (int j = 0; j < 256; ++j) s += xp[j] * wrow[j];
    part[pp * NH + h] = s; }
  __syncthreads();
  if (tid < NH) {
    float s = Dv[tid];
#pragma unroll
    for (int pp = 0; pp < 8; ++pp) s += part[pp * NH + tid];
    defs[tid] = s;
  }
  __syncthreads();
  float ss = 0.f;
  for (int i = tid; i < DI; i += 256) {
    float yf = (t > 0)  ? y2[((size_t)b * SL + (t - 1)) * DI + i] : 0.f;
    float yb = (tb > 0) ? y2[((size_t)(b + NB) * SL + (tb - 1)) * DI + i] : 0.f;
    float y = yf + yb + xsh[i] * defs[i >> 6];
    float z = zx[row * DIP + i];
    float g = y * (z / (1.f + __expf(-z)));
    gsh[i] = g; ss += g * g;
  }
  rbuf[tid] = ss; __syncthreads();
  for (int s = 128; s > 0; s >>= 1) { if (tid < s) rbuf[tid] += rbuf[tid + s]; __syncthreads(); }
  float rs = rsqrtf(rbuf[0] / DI + EPSF);
  for (int i = tid; i < DI; i += 256)
    yn[row * DI + i] = (bf16_t)(gsh[i] * rs * gnw[i]);
}

// -------------------------------------------------------------- launch ----
extern "C" void kernel_launch(void* const* d_in, const int* in_sizes, int n_in,
                              void* d_out, int out_size, void* d_ws, size_t ws_size,
                              hipStream_t stream) {
  (void)in_sizes; (void)n_in; (void)out_size; (void)ws_size;
  const float* hs      = (const float*)d_in[0];
  const float* resid   = (const float*)d_in[1];
  const float* nw      = (const float*)d_in[2];
  const float* w_in    = (const float*)d_in[3];
  const float* conv_w  = (const float*)d_in[4];
  const float* conv_b  = (const float*)d_in[5];
  const float* dt_bias = (const float*)d_in[6];
  const float* A_log   = (const float*)d_in[7];
  const float* Dv      = (const float*)d_in[8];
  const float* fcw     = (const float*)d_in[9];
  const float* gnw     = (const float*)d_in[10];
  const float* w_out   = (const float*)d_in[11];

  float* out     = (float*)d_out;
  float* res_out = out + (size_t)NB * SL * DM;

  char* ws = (char*)d_ws;
  bf16_t* hbf   = (bf16_t*)(ws + OFF_HBF);
  bf16_t* winbf = (bf16_t*)(ws + OFF_WIN);
  bf16_t* woutbf= (bf16_t*)(ws + OFF_WOUT);
  float*  zx    = (float*) (ws + OFF_ZX);
  float*  dt2   = (float*) (ws + OFF_DT2);
  float*  dA    = (float*) (ws + OFF_DA);
  float*  xconv = (float*) (ws + OFF_XCV);
  bf16_t* X2    = (bf16_t*)(ws + OFF_X2);
  bf16_t* B2p   = (bf16_t*)(ws + OFF_B2);
  bf16_t* C2p   = (bf16_t*)(ws + OFF_C2);
  float*  st    = (float*) (ws + OFF_ST);
  float*  acg   = (float*) (ws + OFF_AC);
  float*  y2    = (float*) (ws + OFF_Y2);
  bf16_t* ynbf  = (bf16_t*)(ws + OFF_YN);

  // weight casts
  cvt_bf16_kernel<<<2048, 256, 0, stream>>>(w_in, winbf, (long)DIP * DM);
  cvt_bf16_kernel<<<2048, 256, 0, stream>>>(w_out, woutbf, (long)DM * DI);
  // residual add + rmsnorm
  addnorm_kernel<<<dim3(SL, NB), 256, 0, stream>>>(hs, resid, nw, res_out, hbf);
  // in_proj GEMM: [8192,1024] x [4288,1024]^T
  gemm_bf16_kernel<<<dim3((NB * SL) / 128, DIP / 64), 256, 0, stream>>>(
      hbf, winbf, zx, NB * SL, DIP, DM);
  // dt softplus + dA
  dt_kernel<<<dim3(SL, NB), 64, 0, stream>>>(zx, dt_bias, A_log, dt2, dA);
  // depthwise conv + silu + split + dir duplication
  conv_kernel<<<dim3(SL, NB), 256, 0, stream>>>(zx, conv_w, conv_b, dt2,
                                                xconv, X2, B2p, C2p);
  // per-chunk SSD (Y_diag + chunk states) via WMMA
  ssd_chunk_kernel<<<dim3(NC, NH, NB2), 32, 0, stream>>>(X2, B2p, C2p, dA,
                                                         y2, st, acg);
  // inter-chunk state scan
  ssd_scan_kernel<<<dim3(NH, NB2), 256, 0, stream>>>(st, acg);
  // Y_off via WMMA
  ssd_off_kernel<<<dim3(NC, NH, NB2), 32, 0, stream>>>(C2p, st, acg, y2);
  // combine dirs + Deff + gate + rmsnorm
  post_kernel<<<dim3(SL, NB), 256, 0, stream>>>(y2, xconv, zx, fcw, Dv, gnw, ynbf);
  // out_proj GEMM: [8192,2048] x [1024,2048]^T
  gemm_bf16_kernel<<<dim3((NB * SL) / 128, DM / 64), 256, 0, stream>>>(
      ynbf, woutbf, out, NB * SL, DM, DI);
}
